// LSTMModel_88424786690438
// MI455X (gfx1250) — matrix operand order repaired and checked by reading, not compiled, not run on hardware
//
#include <hip/hip_runtime.h>
#include <hip/hip_bf16.h>
#include <stdint.h>
#include <stddef.h>

// ---------------- types for WMMA ----------------
typedef __attribute__((ext_vector_type(16))) __bf16 v16bf;
typedef __attribute__((ext_vector_type(8)))  __bf16 v8bf;
typedef __attribute__((ext_vector_type(8)))  float  v8f;

#define S_LEN  512
#define BATCH  32
#define HID    1024
#define GATE4  4096
#define SB     (S_LEN * BATCH)          // 16384 rows
#define LAYERW ((size_t)GATE4 * HID)    // elements per layer weight matrix

static __device__ __forceinline__ unsigned short f2bf(float f) {
  union { float f; unsigned u; } v; v.f = f;
  unsigned u = v.u;
  u += 0x7fffu + ((u >> 16) & 1u);      // round-to-nearest-even
  return (unsigned short)(u >> 16);
}
static __device__ __forceinline__ float sigmoidf_(float x) {
  return 1.0f / (1.0f + __expf(-x));
}

// A-tile (16x32 bf16): lanes 0-15 hold rows 0-15, K0..15;
//                      lanes 16-31 hold rows 0-15, K16..31
static __device__ __forceinline__ v16bf load_tileA(const __bf16* __restrict__ base,
                                                   int row, int k0, int koff) {
  const __bf16* p = base + (size_t)row * HID + k0 + koff;
  v8bf lo = *reinterpret_cast<const v8bf*>(p);
  v8bf hi = *reinterpret_cast<const v8bf*>(p + 8);
  return __builtin_shufflevector(lo, hi, 0,1,2,3,4,5,6,7,8,9,10,11,12,13,14,15);
}
// B-tile (32x16 bf16): lanes 0-15 col N K0..15; lanes 16-31 col N K16..31
// (weight rows are exactly B columns: K-contiguous, one 32B load)
static __device__ __forceinline__ v16bf load_tileB(const __bf16* __restrict__ base,
                                                   int col, int k0, int koffb) {
  return *reinterpret_cast<const v16bf*>(base + (size_t)col * HID + k0 + koffb);
}

// ---------------- fp32 -> bf16 weight conversion ----------------
__global__ void cvt_f32_bf16(const float* __restrict__ in,
                             unsigned short* __restrict__ out, long n) {
  long i = (long)blockIdx.x * blockDim.x + threadIdx.x;
  long stride = (long)gridDim.x * blockDim.x;
  for (; i < n; i += stride) out[i] = f2bf(in[i]);
}

// ---------------- embedding gather -> bf16 [SB, HID] ----------------
__global__ void embed_bf16(const int* __restrict__ tok,
                           const float* __restrict__ emb,
                           unsigned short* __restrict__ out) {
  long n = (long)SB * HID;
  long i = (long)blockIdx.x * blockDim.x + threadIdx.x;
  long stride = (long)gridDim.x * blockDim.x;
  for (; i < n; i += stride) {
    int row = (int)(i >> 10);          // HID == 1024
    int h   = (int)(i & 1023);
    out[i] = f2bf(emb[(long)tok[row] * HID + h]);
  }
}

// ---------------- big input-projection GEMM ----------------
// C[SB, GATE4] = A[SB,HID](bf16) * W[GATE4,HID]^T (bf16) + bias
// One wave: 32(M) x 64(N) tile, 8 accumulators, A reused over 4 N-tiles,
// B reused over 2 M-tiles. Load-all-then-compute per k-step.
__global__ __launch_bounds__(256) void gemm_xproj(
    const unsigned short* __restrict__ A,
    const unsigned short* __restrict__ W,
    const float* __restrict__ bias,
    float* __restrict__ C) {
  const int lane   = threadIdx.x & 31;
  const int wid    = blockIdx.x * 8 + (threadIdx.x >> 5);   // 8 waves / block
  const int m_base = (wid >> 6) * 32;                       // 512 M-supertiles
  const int n_base = (wid & 63) * 64;                       // 64 N-groups
  const int row_a  = lane & 15;
  const int koff_a = (lane & 16) ? 16 : 0;
  const int col_b  = lane & 15;
  const int koff_b = (lane & 16) ? 16 : 0;

  const __bf16* Ab = reinterpret_cast<const __bf16*>(A);
  const __bf16* Wb = reinterpret_cast<const __bf16*>(W);

  v8f acc[2][4];
#pragma unroll
  for (int mi = 0; mi < 2; ++mi)
#pragma unroll
    for (int nj = 0; nj < 4; ++nj) acc[mi][nj] = v8f{};

  for (int k0 = 0; k0 < HID; k0 += 32) {
    v16bf a[2], b[4];
#pragma unroll
    for (int mi = 0; mi < 2; ++mi)
      a[mi] = load_tileA(Ab, m_base + mi * 16 + row_a, k0, koff_a);
#pragma unroll
    for (int nj = 0; nj < 4; ++nj)
      b[nj] = load_tileB(Wb, n_base + nj * 16 + col_b, k0, koff_b);
#pragma unroll
    for (int mi = 0; mi < 2; ++mi)
#pragma unroll
      for (int nj = 0; nj < 4; ++nj)
        acc[mi][nj] = __builtin_amdgcn_wmma_f32_16x16x32_bf16(
            false, a[mi], false, b[nj], (short)0, acc[mi][nj], false, false);
  }
  // C/D layout: VGPR r -> (lanes 0-15: M=r, N=lane) (lanes 16-31: M=r+8, N=lane-16)
  const int m_lo = (lane & 16) ? 8 : 0;
#pragma unroll
  for (int nj = 0; nj < 4; ++nj) {
    int n = n_base + nj * 16 + col_b;
    float bs = bias[n];
#pragma unroll
    for (int mi = 0; mi < 2; ++mi)
#pragma unroll
      for (int r = 0; r < 8; ++r) {
        int m = m_base + mi * 16 + m_lo + r;
        C[(size_t)m * GATE4 + n] = acc[mi][nj][r] + bs;
      }
  }
}

// ---------------- h/c state (re)initialization ----------------
__global__ void state_init(float* __restrict__ hF, float* __restrict__ cF,
                           unsigned short* __restrict__ hB,
                           const float* __restrict__ srcH,
                           const float* __restrict__ srcC,
                           unsigned* __restrict__ cnt, int zero) {
  int i = blockIdx.x * blockDim.x + threadIdx.x;   // grid covers BATCH*HID
  float h = zero ? 0.0f : srcH[i];
  float c = zero ? 0.0f : srcC[i];
  hF[i] = h; cF[i] = c; hB[i] = f2bf(h);
  if (i == 0) *cnt = 0u;
}

// ---------------- grid-wide barrier (monotonic counter) ----------------
static __device__ __forceinline__ void grid_sync(unsigned* cnt, unsigned target) {
  __syncthreads();
  if (threadIdx.x == 0) {
    __threadfence();
    atomicAdd(cnt, 1u);
    while (atomicAdd(cnt, 0u) < target) { __builtin_amdgcn_s_sleep(8); }
  }
  __syncthreads();
}

// ---------------- persistent recurrent LSTM kernel ----------------
// 16 blocks x 256 threads = 128 waves; each wave computes the FULL batch
// (M=32) x 32 gate columns, so Whh traffic per step is halved and each wave
// has 4 independent WMMA chains. Double-buffered k-loop + xproj prefetch.
__global__ __launch_bounds__(256) void lstm_recur(
    const float* __restrict__ xproj,          // [S, B, GATE4] fp32 (bias included)
    const unsigned short* __restrict__ Whh,   // [GATE4, HID] bf16
    float* __restrict__ gates,                // [B, GATE4] fp32 scratch
    float* __restrict__ hF, float* __restrict__ cF,   // [B, HID] fp32 state
    unsigned short* __restrict__ hB,          // [B, HID] bf16 state
    unsigned short* __restrict__ seq_out,     // [S, B, HID] bf16
    float* __restrict__ fout,                 // optional [S, B, HID] fp32 (or null)
    unsigned* __restrict__ cnt) {
  const int lane   = threadIdx.x & 31;
  const int wid    = blockIdx.x * 8 + (threadIdx.x >> 5);  // 0..127
  const int n_base = wid * 32;                             // 128 x 32 cols = 4096
  const int row_a  = lane & 15;
  const int koff_a = (lane & 16) ? 16 : 0;
  const int col_b  = lane & 15;
  const int koff_b = (lane & 16) ? 16 : 0;
  const int m_lo   = (lane & 16) ? 8 : 0;
  const int tid_g  = blockIdx.x * 256 + threadIdx.x;       // 0..4095
  const unsigned nblk = gridDim.x;

  const __bf16* Wb  = reinterpret_cast<const __bf16*>(Whh);
  const __bf16* hBb = reinterpret_cast<const __bf16*>(hB);

  unsigned bar = 0;
  for (int t = 0; t < S_LEN; ++t) {
    const float* xp = xproj + (size_t)t * BATCH * GATE4;
    // Prefetch this step's xproj contribution: the 32 scattered HBM loads
    // are issued now and complete behind the 32-iteration matmul.
    float xr[2][2][8];
#pragma unroll
    for (int nj = 0; nj < 2; ++nj) {
      int n = n_base + nj * 16 + col_b;
#pragma unroll
      for (int mi = 0; mi < 2; ++mi)
#pragma unroll
        for (int r = 0; r < 8; ++r)
          xr[mi][nj][r] = xp[(size_t)(mi * 16 + m_lo + r) * GATE4 + n];
    }

    // ---- phase A: g_tile = hB @ Whh^T (double-buffered WMMA loop) ----
    v8f acc[2][2];
#pragma unroll
    for (int mi = 0; mi < 2; ++mi)
#pragma unroll
      for (int nj = 0; nj < 2; ++nj) acc[mi][nj] = v8f{};

    v16bf a[2], b[2];
#pragma unroll
    for (int mi = 0; mi < 2; ++mi) a[mi] = load_tileA(hBb, mi * 16 + row_a, 0, koff_a);
#pragma unroll
    for (int nj = 0; nj < 2; ++nj) b[nj] = load_tileB(Wb, n_base + nj * 16 + col_b, 0, koff_b);

    for (int k0 = 0; k0 < HID - 32; k0 += 32) {
      v16bf an[2], bn[2];
      int kn = k0 + 32;
#pragma unroll
      for (int mi = 0; mi < 2; ++mi) an[mi] = load_tileA(hBb, mi * 16 + row_a, kn, koff_a);
#pragma unroll
      for (int nj = 0; nj < 2; ++nj) bn[nj] = load_tileB(Wb, n_base + nj * 16 + col_b, kn, koff_b);
#pragma unroll
      for (int mi = 0; mi < 2; ++mi)
#pragma unroll
        for (int nj = 0; nj < 2; ++nj)
          acc[mi][nj] = __builtin_amdgcn_wmma_f32_16x16x32_bf16(
              false, a[mi], false, b[nj], (short)0, acc[mi][nj], false, false);
#pragma unroll
      for (int mi = 0; mi < 2; ++mi) a[mi] = an[mi];
#pragma unroll
      for (int nj = 0; nj < 2; ++nj) b[nj] = bn[nj];
    }
#pragma unroll
    for (int mi = 0; mi < 2; ++mi)
#pragma unroll
      for (int nj = 0; nj < 2; ++nj)
        acc[mi][nj] = __builtin_amdgcn_wmma_f32_16x16x32_bf16(
            false, a[mi], false, b[nj], (short)0, acc[mi][nj], false, false);

#pragma unroll
    for (int nj = 0; nj < 2; ++nj) {
      int n = n_base + nj * 16 + col_b;
#pragma unroll
      for (int mi = 0; mi < 2; ++mi)
#pragma unroll
        for (int r = 0; r < 8; ++r) {
          int m = mi * 16 + m_lo + r;
          gates[(size_t)m * GATE4 + n] = acc[mi][nj][r] + xr[mi][nj][r];
        }
    }
    ++bar; grid_sync(cnt, bar * nblk);

    // ---- phase B: elementwise LSTM cell update (i,f,g,o) ----
#pragma unroll
    for (int e = 0; e < (BATCH * HID) / 4096; ++e) {       // 8 elems / thread
      int idx = e * 4096 + tid_g;                          // coalesced
      int bb = idx >> 10, h = idx & 1023;
      const float* gr = gates + (size_t)bb * GATE4;
      float gi = gr[h];
      float gf = gr[h + 1024];
      float gg = gr[h + 2048];
      float go = gr[h + 3072];
      float c  = cF[idx];
      float cn = sigmoidf_(gf) * c + sigmoidf_(gi) * tanhf(gg);
      float hn = sigmoidf_(go) * tanhf(cn);
      cF[idx] = cn;
      hF[idx] = hn;
      hB[idx] = f2bf(hn);
      size_t so = (size_t)t * BATCH * HID + idx;
      seq_out[so] = f2bf(hn);
      if (fout) fout[so] = hn;
    }
    ++bar; grid_sync(cnt, bar * nblk);
  }
}

// ---------------- host launcher ----------------
extern "C" void kernel_launch(void* const* d_in, const int* in_sizes, int n_in,
                              void* d_out, int out_size, void* d_ws, size_t ws_size,
                              hipStream_t stream) {
  const int*   x       = (const int*)d_in[0];
  const float* emb_enc = (const float*)d_in[1];
  const float* enc_Wih = (const float*)d_in[2];
  const float* enc_Whh = (const float*)d_in[3];
  const float* enc_b   = (const float*)d_in[4];
  const float* emb_dec = (const float*)d_in[5];
  const float* dec_Wih = (const float*)d_in[6];
  const float* dec_Whh = (const float*)d_in[7];
  const float* dec_b   = (const float*)d_in[8];
  float* out = (float*)d_out;

  char* ws = (char*)d_ws;
  size_t off = 0;
  auto alloc = [&](size_t bytes) -> char* {
    char* p = ws + off;
    off = (off + bytes + 255) & ~(size_t)255;
    return p;
  };
  unsigned short* wih_bf = (unsigned short*)alloc(6 * LAYERW * 2);   // enc0-2, dec0-2
  unsigned short* whh_bf = (unsigned short*)alloc(6 * LAYERW * 2);
  unsigned short* seqA   = (unsigned short*)alloc((size_t)SB * HID * 2);
  unsigned short* seqB   = (unsigned short*)alloc((size_t)SB * HID * 2);
  float* xproj = (float*)alloc((size_t)SB * GATE4 * 4);
  float* gates = (float*)alloc((size_t)BATCH * GATE4 * 4);
  float* hF    = (float*)alloc((size_t)BATCH * HID * 4);
  float* cF    = (float*)alloc((size_t)BATCH * HID * 4);
  unsigned short* hB = (unsigned short*)alloc((size_t)BATCH * HID * 2);
  float* finH  = (float*)alloc((size_t)3 * BATCH * HID * 4);
  float* finC  = (float*)alloc((size_t)3 * BATCH * HID * 4);
  unsigned* cnt = (unsigned*)alloc(64);

  const long wn = 3 * (long)LAYERW;
  cvt_f32_bf16<<<1024, 256, 0, stream>>>(enc_Wih, wih_bf,      wn);
  cvt_f32_bf16<<<1024, 256, 0, stream>>>(dec_Wih, wih_bf + wn, wn);
  cvt_f32_bf16<<<1024, 256, 0, stream>>>(enc_Whh, whh_bf,      wn);
  cvt_f32_bf16<<<1024, 256, 0, stream>>>(dec_Whh, whh_bf + wn, wn);

  const int BH = BATCH * HID;
  unsigned short* sin = seqA;
  unsigned short* sout = seqB;

  // ---- encoder: zero initial state, save finals per layer ----
  embed_bf16<<<2048, 256, 0, stream>>>(x, emb_enc, seqA);
  for (int l = 0; l < 3; ++l) {
    gemm_xproj<<<4096, 256, 0, stream>>>(sin, wih_bf + (size_t)l * LAYERW,
                                         enc_b + l * GATE4, xproj);
    state_init<<<128, 256, 0, stream>>>(hF, cF, hB, finH, finC, cnt, 1);
    lstm_recur<<<16, 256, 0, stream>>>(xproj, whh_bf + (size_t)l * LAYERW,
                                       gates, hF, cF, hB, sout, nullptr, cnt);
    hipMemcpyAsync(finH + (size_t)l * BH, hF, (size_t)BH * 4,
                   hipMemcpyDeviceToDevice, stream);
    hipMemcpyAsync(finC + (size_t)l * BH, cF, (size_t)BH * 4,
                   hipMemcpyDeviceToDevice, stream);
    unsigned short* t = sin; sin = sout; sout = t;
  }

  // ---- decoder: init from encoder finals; top layer writes fp32 output ----
  sin = seqA; sout = seqB;
  embed_bf16<<<2048, 256, 0, stream>>>(x, emb_dec, seqA);
  for (int l = 0; l < 3; ++l) {
    gemm_xproj<<<4096, 256, 0, stream>>>(sin, wih_bf + (size_t)(3 + l) * LAYERW,
                                         dec_b + l * GATE4, xproj);
    state_init<<<128, 256, 0, stream>>>(hF, cF, hB, finH + (size_t)l * BH,
                                        finC + (size_t)l * BH, cnt, 0);
    lstm_recur<<<16, 256, 0, stream>>>(xproj, whh_bf + (size_t)(3 + l) * LAYERW,
                                       gates, hF, cF, hB, sout,
                                       (l == 2) ? out : nullptr, cnt);
    unsigned short* t = sin; sin = sout; sout = t;
  }
}
